// lstm_49143015801508
// MI455X (gfx1250) — matrix operand order repaired and checked by reading, not compiled, not run on hardware
//
#include <hip/hip_runtime.h>
#include <hip/hip_bf16.h>

// Problem constants
#define B_   64
#define T_   512
#define F_   20
#define D_   1024
#define O_   1095
#define L_   3
#define M_   (B_ * T_)     // 32768 rows (time-major: m = t*B + b)
#define N3D_ 3072          // 3*D
#define OPAD 1152          // O padded to multiple of 128

#define LDSS 40            // LDS tile stride in shorts (80B = 16B aligned rows)

typedef __attribute__((ext_vector_type(16))) __bf16 v16bf;
typedef __attribute__((ext_vector_type(8)))  float  v8f;

union FragBF { unsigned int u[8]; v16bf v; };

__device__ __forceinline__ unsigned short f2bf(float f) {
    unsigned int u = __float_as_uint(f);
    u += 0x7FFFu + ((u >> 16) & 1u);   // round-to-nearest-even
    return (unsigned short)(u >> 16);
}
__device__ __forceinline__ float bf2f(unsigned short h) {
    return __uint_as_float(((unsigned int)h) << 16);
}

// Generic-pointer low 32 bits == LDS offset for shared objects on AMDGPU.
__device__ __forceinline__ unsigned lds_off(const void* p) {
    return (unsigned)(size_t)p;
}
// Async DMA: copy 16B per lane from global to LDS (ASYNCcnt-tracked).
__device__ __forceinline__ void async_cp16(unsigned lds_addr, const void* gptr) {
    asm volatile("global_load_async_to_lds_b128 %0, %1, off"
                 :: "v"(lds_addr), "v"(gptr) : "memory");
}

// ---------------------------------------------------------------------------
// Tiled transpose + f32->bf16 convert: src (Ksrc x Nsrc f32, row-major) ->
// dst (Ndst x Ksrc bf16, row-major), zero-padding rows n >= Nsrc.
// ---------------------------------------------------------------------------
__global__ __launch_bounds__(256) void transpose_cvt(
    const float* __restrict__ src, unsigned short* __restrict__ dst,
    int Nsrc, int Ksrc)
{
    __shared__ float tile[32][33];
    const int bn = blockIdx.x * 32;   // n tile (dst row block)
    const int bk = blockIdx.y * 32;   // k tile
    const int r = threadIdx.x >> 5, c = threadIdx.x & 31;
#pragma unroll
    for (int i = 0; i < 4; ++i) {
        const int kk = bk + r + 8 * i;
        const int nn = bn + c;
        tile[r + 8 * i][c] = (nn < Nsrc) ? src[(size_t)kk * Nsrc + nn] : 0.0f;
    }
    __syncthreads();
#pragma unroll
    for (int i = 0; i < 4; ++i) {
        const int nn = bn + r + 8 * i;
        dst[(size_t)nn * Ksrc + bk + c] = f2bf(tile[c][r + 8 * i]);
    }
}

// ---------------------------------------------------------------------------
// Input projection: h = x @ W1 + b1, stored time-major (t,b,d) as bf16
// ---------------------------------------------------------------------------
__global__ __launch_bounds__(256) void input_proj(
    const float* __restrict__ x, const float* __restrict__ W1,
    const float* __restrict__ b1, unsigned short* __restrict__ HB)
{
    size_t idx = (size_t)blockIdx.x * 256 + threadIdx.x;  // < M_*D_
    int    d = (int)(idx & (D_ - 1));
    size_t m = idx >> 10;              // time-major row: t*B + b
    int    t = (int)(m >> 6);
    int    b = (int)(m & 63);
    const float* xr = x + ((size_t)b * T_ + t) * F_;
    float s = b1[d];
#pragma unroll
    for (int f = 0; f < F_; ++f)
        s += xr[f] * W1[(size_t)f * D_ + d];
    HB[idx] = f2bf(s);
}

// ---------------------------------------------------------------------------
// bf16 WMMA GEMM: C(MxN,f32) = A(MxK bf16 rowmajor) * Bt(Npad x K bf16)^T
// Block 128x128, 8 waves (4x2), wave tile 32x64, K-step 32.
// Double-buffered LDS filled by GLOBAL_LOAD_ASYNC_TO_LDS_B128 (ASYNCcnt).
// mode 0: plain store (ld=Npad).  mode 1: +bias, n<Nstore mask, row remap.
// ---------------------------------------------------------------------------
__global__ __launch_bounds__(256) void gemm_bf16_wmma(
    const unsigned short* __restrict__ A,
    const unsigned short* __restrict__ Bt,
    float* __restrict__ Cout,
    const float* __restrict__ bias,
    int K, int Npad, int Nstore, int mode)
{
    __shared__ unsigned short As[2][128 * LDSS];
    __shared__ unsigned short Bs[2][128 * LDSS];

    const int tid  = threadIdx.x;
    const int lane = tid & 31;
    const int wv   = tid >> 5;
    const int wm   = wv >> 1;            // 0..3
    const int wn   = wv & 1;             // 0..1
    const int m0   = blockIdx.y * 128;
    const int n0   = blockIdx.x * 128;

    // Tile copy mapping: A and B tiles are both 128 rows x 32 k (bf16).
    // 512 16B-chunks per tile; this thread owns rows r0, r1 at k-chunk kc.
    const int r0 = tid >> 2;             // 0..63
    const int r1 = r0 + 64;              // 64..127
    const int kc = (tid & 3) * 8;        // 0,8,16,24

    const unsigned short* Ag0 = A  + (size_t)(m0 + r0) * K + kc;
    const unsigned short* Ag1 = A  + (size_t)(m0 + r1) * K + kc;
    const unsigned short* Bg0 = Bt + (size_t)(n0 + r0) * K + kc;
    const unsigned short* Bg1 = Bt + (size_t)(n0 + r1) * K + kc;

    const unsigned lA[2] = { lds_off(&As[0][r0 * LDSS + kc]), lds_off(&As[1][r0 * LDSS + kc]) };
    const unsigned lA2[2]= { lds_off(&As[0][r1 * LDSS + kc]), lds_off(&As[1][r1 * LDSS + kc]) };
    const unsigned lB[2] = { lds_off(&Bs[0][r0 * LDSS + kc]), lds_off(&Bs[1][r0 * LDSS + kc]) };
    const unsigned lB2[2]= { lds_off(&Bs[0][r1 * LDSS + kc]), lds_off(&Bs[1][r1 * LDSS + kc]) };

    v8f acc[2][4];
#pragma unroll
    for (int i = 0; i < 2; ++i)
#pragma unroll
        for (int j = 0; j < 4; ++j) acc[i][j] = (v8f)0.0f;

    const int lr = lane & 15;   // row/col within 16
    const int lg = lane >> 4;   // K-half selector

    // Preload tile 0 into buffer 0
    async_cp16(lA[0],  Ag0);
    async_cp16(lA2[0], Ag1);
    async_cp16(lB[0],  Bg0);
    async_cp16(lB2[0], Bg1);

    for (int kt = 0; kt < K; kt += 32) {
        const int cur = (kt >> 5) & 1;
        if (kt + 32 < K) {
            const int nxt = cur ^ 1;
            async_cp16(lA[nxt],  Ag0 + kt + 32);
            async_cp16(lA2[nxt], Ag1 + kt + 32);
            async_cp16(lB[nxt],  Bg0 + kt + 32);
            async_cp16(lB2[nxt], Bg1 + kt + 32);
            if (kt + 64 < K) {   // warm L2 two tiles ahead for the DMA
                __builtin_prefetch(Ag0 + kt + 64, 0, 1);
                __builtin_prefetch(Bg0 + kt + 64, 0, 1);
            }
            // 4 newly issued may remain outstanding; previous 4 are done.
            asm volatile("s_wait_asynccnt 0x4" ::: "memory");
        } else {
            asm volatile("s_wait_asynccnt 0x0" ::: "memory");
        }
        __syncthreads();   // current buffer fully resident for all waves

        // A fragments (16x32 bf16): lane=M, dword i holds K pair per ISA layout
        FragBF af[2];
#pragma unroll
        for (int mf = 0; mf < 2; ++mf) {
            const int row = wm * 32 + mf * 16 + lr;
#pragma unroll
            for (int i = 0; i < 8; ++i) {
                const int k = (i < 4) ? (lg * 8 + 2 * i) : (16 + lg * 8 + 2 * (i - 4));
                af[mf].u[i] = *(const unsigned int*)&As[cur][row * LDSS + k];
            }
        }
        // B fragments (32x16): lane=N, K pairs contiguous (Bt rows are K-major)
#pragma unroll
        for (int nf = 0; nf < 4; ++nf) {
            FragBF bfr;
            const int col = wn * 64 + nf * 16 + lr;
#pragma unroll
            for (int i = 0; i < 8; ++i) {
                const int k = (i < 4) ? (lg * 8 + 2 * i) : (16 + lg * 8 + 2 * (i - 4));
                bfr.u[i] = *(const unsigned int*)&Bs[cur][col * LDSS + k];
            }
            acc[0][nf] = __builtin_amdgcn_wmma_f32_16x16x32_bf16(
                false, af[0].v, false, bfr.v, (short)0, acc[0][nf], false, false);
            acc[1][nf] = __builtin_amdgcn_wmma_f32_16x16x32_bf16(
                false, af[1].v, false, bfr.v, (short)0, acc[1][nf], false, false);
        }
        __syncthreads();   // release 'cur' before it is overwritten by DMA
    }

    // Store: C/D layout -> VGPR r: M = r + 8*lg, N = lane&15
#pragma unroll
    for (int mf = 0; mf < 2; ++mf) {
#pragma unroll
        for (int nf = 0; nf < 4; ++nf) {
            const int nc = n0 + wn * 64 + nf * 16 + lr;
#pragma unroll
            for (int r = 0; r < 8; ++r) {
                const int mr = m0 + wm * 32 + mf * 16 + r + 8 * lg;
                const float val = acc[mf][nf][r];
                if (mode == 0) {
                    Cout[(size_t)mr * Npad + nc] = val;
                } else if (nc < Nstore) {
                    const int t = mr >> 6;       // B_ = 64
                    const int b = mr & 63;
                    Cout[((size_t)b * T_ + t) * Nstore + nc] = val + bias[nc];
                }
            }
        }
    }
}

// ---------------------------------------------------------------------------
// SRU recurrence: one thread per (b,d), c in register, T sequential steps.
// ---------------------------------------------------------------------------
__global__ __launch_bounds__(256) void sru_scan(
    const float* __restrict__ U, unsigned short* __restrict__ HB,
    const float* __restrict__ v2, const float* __restrict__ b2,
    float* __restrict__ cfin)
{
    const int idx = blockIdx.x * 256 + threadIdx.x;  // < B_*D_
    const int d = idx & (D_ - 1);
    const int b = idx >> 10;
    const float vf  = v2[d],      vr  = v2[D_ + d];
    const float bfv = b2[d],      brv = b2[D_ + d];
    float c = 0.0f;
    for (int t = 0; t < T_; ++t) {
        const size_t row = (size_t)t * B_ + b;
        const size_t ub  = row * N3D_;
        const float u0 = U[ub + d];
        const float u1 = U[ub + D_ + d];
        const float u2 = U[ub + 2 * D_ + d];
        const size_t ho = row * D_ + d;
        const float xt = bf2f(HB[ho]);
        const float f = 1.0f / (1.0f + __expf(-(u1 + vf * c + bfv)));
        c = f * c + (1.0f - f) * u0;
        const float r = 1.0f / (1.0f + __expf(-(u2 + vr * c + brv)));
        HB[ho] = f2bf(r * c + (1.0f - r) * xt);
    }
    cfin[(size_t)b * D_ + d] = c;
}

// ---------------------------------------------------------------------------
extern "C" void kernel_launch(void* const* d_in, const int* in_sizes, int n_in,
                              void* d_out, int out_size, void* d_ws, size_t ws_size,
                              hipStream_t stream)
{
    const float* x    = (const float*)d_in[0];
    // d_in[1] = c (ignored by the reference)
    const float* W1   = (const float*)d_in[2];
    const float* b1   = (const float*)d_in[3];
    const float* sruW = (const float*)d_in[4];
    const float* sruV = (const float*)d_in[5];
    const float* sruB = (const float*)d_in[6];
    const float* W3   = (const float*)d_in[7];
    const float* b3   = (const float*)d_in[8];
    float* out = (float*)d_out;

    // Workspace layout (bytes), all 256-aligned:
    //   U   : M_*3072 f32                    = 402,653,184
    //   HB  : M_*1024 bf16                   =  67,108,864
    //   WBt : 3 x [3072][1024] bf16 (W^T)    =  18,874,368
    //   W3t : [1152][1024] bf16 (W3^T, pad)  =   2,359,296
    char* ws = (char*)d_ws;
    float*          U   = (float*)ws;
    unsigned short* HB  = (unsigned short*)(ws + (size_t)402653184);
    unsigned short* WBt = (unsigned short*)(ws + (size_t)469762048);
    unsigned short* W3t = (unsigned short*)(ws + (size_t)488636416);

    // 1. bf16 transposed weight conversion
    for (int l = 0; l < L_; ++l)
        transpose_cvt<<<dim3(N3D_ / 32, D_ / 32), 256, 0, stream>>>(
            sruW + (size_t)l * D_ * N3D_, WBt + (size_t)l * N3D_ * D_, N3D_, D_);
    transpose_cvt<<<dim3(OPAD / 32, D_ / 32), 256, 0, stream>>>(W3, W3t, O_, D_);

    // 2. input projection -> h (time-major bf16)
    input_proj<<<(unsigned)(((size_t)M_ * D_) / 256), 256, 0, stream>>>(x, W1, b1, HB);

    // 3. SRU layers: U = h @ W_l  (WMMA bf16 + async-LDS DMA), then scan
    for (int l = 0; l < L_; ++l) {
        gemm_bf16_wmma<<<dim3(N3D_ / 128, M_ / 128), 256, 0, stream>>>(
            HB, WBt + (size_t)l * N3D_ * D_, U, nullptr, D_, N3D_, N3D_, 0);
        sru_scan<<<(B_ * D_) / 256, 256, 0, stream>>>(
            U, HB,
            sruV + (size_t)l * 2 * D_, sruB + (size_t)l * 2 * D_,
            out + (size_t)M_ * O_ + (size_t)l * B_ * D_);
    }

    // 4. y = h @ W3 + b3, with time-major -> (b*T+t) row remap, n<O mask
    gemm_bf16_wmma<<<dim3(OPAD / 128, M_ / 128), 256, 0, stream>>>(
        HB, W3t, out, b3, D_, OPAD, O_, 1);
}
